// GraphTransformerEncoder_90855738179986
// MI455X (gfx1250) — matrix operand order repaired and checked by reading, not compiled, not run on hardware
//
#include <hip/hip_runtime.h>
#include <hip/hip_bf16.h>

#define DD   128
#define HH   4
#define GG   64
#define OUTD 10
#define LL   3

typedef __attribute__((ext_vector_type(16))) __bf16 v16bf;
typedef __attribute__((ext_vector_type(8)))  float  v8f;

union FragBF {
    v16bf v;
    uint4 q[2];
};

__device__ __forceinline__ unsigned short f32_to_bf16_rne(float f) {
    unsigned u = __float_as_uint(f);
    unsigned r = u + 0x7FFFu + ((u >> 16) & 1u);
    return (unsigned short)(r >> 16);
}

// ---------------- conversion kernels ----------------

__global__ void cvt_f32_bf16_kernel(const float* __restrict__ in,
                                    unsigned short* __restrict__ out, int n) {
    int i = blockIdx.x * blockDim.x + threadIdx.x;
    if (i < n) out[i] = f32_to_bf16_rne(in[i]);
}

// W [Din, Nout] row-major fp32 -> Wt [Nout, Din] bf16
__global__ void cvt_transpose_w_kernel(const float* __restrict__ W,
                                       unsigned short* __restrict__ Wt,
                                       int Din, int Nout) {
    int i = blockIdx.x * blockDim.x + threadIdx.x;
    if (i >= Din * Nout) return;
    int k = i / Nout;
    int n = i - k * Nout;
    Wt[n * Din + k] = f32_to_bf16_rne(W[i]);
}

// ---------------- WMMA GEMM ----------------
// C[M,Nout] = A[M,128](bf16) * W(bf16, pre-transposed [Nout,128]) + bias
// One wave computes a 16x64 strip (4 fp32 accumulators). Per k-chunk all
// five fragments (1xA + 4xB) are live at once so the 10 b128 loads clause
// together and the WMMAs only take partial loadcnt waits.

__global__ __launch_bounds__(256) void gemm_bf16_wmma_kernel(
    const unsigned short* __restrict__ A,    // [M,128] bf16 row-major
    const unsigned short* __restrict__ Bt,   // [Nout,128] bf16 (transposed weight)
    const float* __restrict__ bias,          // [Nout]
    float* __restrict__ C,                   // [M,Nout] fp32
    int M, int Nout) {
    const int lane    = threadIdx.x & 31;
    const int wave    = blockIdx.x * (blockDim.x >> 5) + (threadIdx.x >> 5);
    const int groupsN = Nout >> 6;           // 64-wide column groups
    const int tilesM  = M >> 4;
    if (wave >= tilesM * groupsN) return;
    const int tm = wave / groupsN;
    const int tg = wave - tm * groupsN;

    const int l15 = lane & 15;
    const int hi  = lane >> 4;               // 0: lanes 0-15, 1: lanes 16-31
    const int koffA = hi * 16;               // A k order used here (same as B): halves hold K {0..15} / {16..31}
    const int koffB = hi * 16;               // B k order used here (same as A): halves hold K {0..15} / {16..31}

    const unsigned short* ap = A + (size_t)(tm * 16 + l15) * DD + koffA;
    const unsigned short* bp0 = Bt + (size_t)(tg * 64 + l15) * DD + koffB;
    const unsigned short* bp1 = bp0 + (size_t)16 * DD;
    const unsigned short* bp2 = bp0 + (size_t)32 * DD;
    const unsigned short* bp3 = bp0 + (size_t)48 * DD;

    v8f acc0 = {}, acc1 = {}, acc2 = {}, acc3 = {};
#pragma unroll
    for (int kb = 0; kb < DD; kb += 32) {
        FragBF a, b0, b1, b2, b3;
        a.q[0]  = *reinterpret_cast<const uint4*>(ap + kb);
        a.q[1]  = *reinterpret_cast<const uint4*>(ap + kb + 8);
        b0.q[0] = *reinterpret_cast<const uint4*>(bp0 + kb);
        b0.q[1] = *reinterpret_cast<const uint4*>(bp0 + kb + 8);
        b1.q[0] = *reinterpret_cast<const uint4*>(bp1 + kb);
        b1.q[1] = *reinterpret_cast<const uint4*>(bp1 + kb + 8);
        b2.q[0] = *reinterpret_cast<const uint4*>(bp2 + kb);
        b2.q[1] = *reinterpret_cast<const uint4*>(bp2 + kb + 8);
        b3.q[0] = *reinterpret_cast<const uint4*>(bp3 + kb);
        b3.q[1] = *reinterpret_cast<const uint4*>(bp3 + kb + 8);
        acc0 = __builtin_amdgcn_wmma_f32_16x16x32_bf16(false, a.v, false, b0.v, (short)0, acc0, false, false);
        acc1 = __builtin_amdgcn_wmma_f32_16x16x32_bf16(false, a.v, false, b1.v, (short)0, acc1, false, false);
        acc2 = __builtin_amdgcn_wmma_f32_16x16x32_bf16(false, a.v, false, b2.v, (short)0, acc2, false, false);
        acc3 = __builtin_amdgcn_wmma_f32_16x16x32_bf16(false, a.v, false, b3.v, (short)0, acc3, false, false);
    }

    const int mrow = tm * 16 + hi * 8;       // C/D layout: VGPR r -> M=r (lanes<16), M=r+8 (lanes>=16)
    const int nb = tg * 64 + l15;
    const float bv0 = bias ? bias[nb]      : 0.0f;
    const float bv1 = bias ? bias[nb + 16] : 0.0f;
    const float bv2 = bias ? bias[nb + 32] : 0.0f;
    const float bv3 = bias ? bias[nb + 48] : 0.0f;
    float* crow = C + (size_t)mrow * Nout + nb;
#pragma unroll
    for (int r = 0; r < 8; ++r) {
        float* cr = crow + (size_t)r * Nout;
        cr[0]  = acc0[r] + bv0;
        cr[16] = acc1[r] + bv1;
        cr[32] = acc2[r] + bv2;
        cr[48] = acc3[r] + bv3;
    }
}

// ---------------- per-layer init ----------------

__global__ void init_layer_kernel(unsigned* __restrict__ m_enc,
                                  float* __restrict__ denom,
                                  float* __restrict__ agg,
                                  int nh, int aggN) {
    int i = blockIdx.x * blockDim.x + threadIdx.x;
    if (i < nh) { m_enc[i] = 0u; denom[i] = 1e-16f; }
    if (i < aggN) agg[i] = 0.0f;
}

// ---------------- edge attention ----------------

// one wave per edge; 4 heads; q,k fp32 [N, H*D]
__global__ __launch_bounds__(256) void edge_scores_kernel(
    const float* __restrict__ q, const float* __restrict__ k,
    const int* __restrict__ src, const int* __restrict__ dst,
    float* __restrict__ scores, int E) {
    int wave = blockIdx.x * (blockDim.x >> 5) + (threadIdx.x >> 5);
    int lane = threadIdx.x & 31;
    if (wave >= E) return;
    int s = src[wave], d = dst[wave];
    const float* qd = q + (size_t)d * (HH * DD);
    const float* ks = k + (size_t)s * (HH * DD);
    __builtin_prefetch(qd + lane * 16, 0, 0);   // global_prefetch_b8: pull both rows toward L2/L0
    __builtin_prefetch(ks + lane * 16, 0, 0);
#pragma unroll
    for (int h = 0; h < HH; ++h) {
        float acc = 0.0f;
#pragma unroll
        for (int j = 0; j < DD; j += 32) {
            int idx = h * DD + j + lane;
            acc += qd[idx] * ks[idx];
        }
#pragma unroll
        for (int off = 16; off > 0; off >>= 1) acc += __shfl_xor(acc, off, 32);
        if (lane == 0) scores[(size_t)wave * HH + h] = acc * 0.08838834764831845f; // 1/sqrt(128)
    }
}

// segment max via monotonic uint encoding + atomicMax
__global__ void edge_segmax_kernel(const float* __restrict__ scores,
                                   const int* __restrict__ dst,
                                   unsigned* __restrict__ m_enc, int EH) {
    int i = blockIdx.x * blockDim.x + threadIdx.x;
    if (i >= EH) return;
    float s = scores[i];
    unsigned u = __float_as_uint(s);
    unsigned enc = (u & 0x80000000u) ? ~u : (u | 0x80000000u);
    atomicMax(&m_enc[dst[i >> 2] * HH + (i & 3)], enc);
}

__global__ void edge_exp_kernel(float* __restrict__ scores,
                                const int* __restrict__ dst,
                                const unsigned* __restrict__ m_enc,
                                float* __restrict__ denom, int EH) {
    int i = blockIdx.x * blockDim.x + threadIdx.x;
    if (i >= EH) return;
    int nh = dst[i >> 2] * HH + (i & 3);
    unsigned enc = m_enc[nh];
    float m;
    if (enc == 0u) {
        m = 0.0f;  // node with no in-edges (matches reference's where(isfinite))
    } else {
        unsigned u = (enc & 0x80000000u) ? (enc & 0x7FFFFFFFu) : ~enc;
        m = __uint_as_float(u);
    }
    float a = __expf(scores[i] - m);
    scores[i] = a;
    atomicAdd(&denom[nh], a);
}

// one wave per (edge, head): agg[dst] += alpha * v[src], 128 coalesced float atomics
__global__ __launch_bounds__(256) void edge_aggregate_kernel(
    const float* __restrict__ scores, const float* __restrict__ denom,
    const float* __restrict__ v,
    const int* __restrict__ src, const int* __restrict__ dst,
    float* __restrict__ agg, int E) {
    int wave = blockIdx.x * (blockDim.x >> 5) + (threadIdx.x >> 5);
    int lane = threadIdx.x & 31;
    if (wave >= E * HH) return;
    int e = wave >> 2, h = wave & 3;
    int s = src[e], d = dst[e];
    const float* vp = v + (size_t)s * (HH * DD) + h * DD;
    __builtin_prefetch(vp + lane * 4, 0, 0);    // hide the random v[src] gather
    float alpha = scores[wave] / denom[d * HH + h];
    float* op = agg + (size_t)d * (HH * DD) + h * DD;
#pragma unroll
    for (int j = 0; j < DD; j += 32) atomicAdd(&op[j + lane], alpha * vp[j + lane]);
}

// mean over heads + skip + relu; emit fp32 and bf16
__global__ void finalize_layer_kernel(const float* __restrict__ agg,
                                      const float* __restrict__ skip,
                                      float* __restrict__ h,
                                      unsigned short* __restrict__ hb, int ND) {
    int i = blockIdx.x * blockDim.x + threadIdx.x;
    if (i >= ND) return;
    int n = i >> 7, dd = i & (DD - 1);
    const float* ap = agg + (size_t)n * (HH * DD) + dd;
    float a = (ap[0] + ap[DD] + ap[2 * DD] + ap[3 * DD]) * 0.25f + skip[i];
    a = fmaxf(a, 0.0f);
    h[i] = a;
    hb[i] = f32_to_bf16_rne(a);
}

// ---------------- pooling + classifier ----------------

__global__ void pool_init_kernel(float* __restrict__ pooled, float* __restrict__ cnt) {
    int i = blockIdx.x * blockDim.x + threadIdx.x;
    if (i < GG * DD) pooled[i] = 0.0f;
    if (i < GG) cnt[i] = 0.0f;
}

__global__ void pool_kernel(const float* __restrict__ h, const int* __restrict__ batch,
                            float* __restrict__ pooled, float* __restrict__ cnt, int ND) {
    int i = blockIdx.x * blockDim.x + threadIdx.x;
    if (i >= ND) return;
    int n = i >> 7, dd = i & (DD - 1);
    int g = batch[n];
    atomicAdd(&pooled[g * DD + dd], h[i]);
    if (dd == 0) atomicAdd(&cnt[g], 1.0f);
}

__global__ void classifier_kernel(const float* __restrict__ pooled, const float* __restrict__ cnt,
                                  const float* __restrict__ Wc, const float* __restrict__ bc,
                                  float* __restrict__ out) {
    int i = blockIdx.x * blockDim.x + threadIdx.x;
    if (i >= GG * OUTD) return;
    int g = i / OUTD, o = i - g * OUTD;
    float inv = 1.0f / fmaxf(cnt[g], 1.0f);
    float acc = bc[o];
    for (int dd = 0; dd < DD; ++dd) acc += pooled[g * DD + dd] * inv * Wc[dd * OUTD + o];
    out[i] = acc;
}

// ---------------- host launcher ----------------

static inline int cdiv(long long a, int b) { return (int)((a + b - 1) / b); }

extern "C" void kernel_launch(void* const* d_in, const int* in_sizes, int n_in,
                              void* d_out, int out_size, void* d_ws, size_t ws_size,
                              hipStream_t stream) {
    (void)n_in; (void)out_size; (void)ws_size;

    const float* x      = (const float*)d_in[0];
    const int*   eidx   = (const int*)d_in[1];
    const int*   batch  = (const int*)d_in[2];
    const float* Wq     = (const float*)d_in[3];
    const float* bq     = (const float*)d_in[4];
    const float* Wk     = (const float*)d_in[5];
    const float* bk     = (const float*)d_in[6];
    const float* Wv     = (const float*)d_in[7];
    const float* bv     = (const float*)d_in[8];
    const float* Wskip  = (const float*)d_in[9];
    const float* bskip  = (const float*)d_in[10];
    const float* Wc     = (const float*)d_in[11];
    const float* bc     = (const float*)d_in[12];
    float* out          = (float*)d_out;

    const int N = in_sizes[0] / DD;
    const int E = in_sizes[1] / 2;
    const int* src = eidx;
    const int* dst = eidx + E;
    const int HD = HH * DD;       // 512
    const int ND = N * DD;
    const int NH = N * HH;
    const int EH = E * HH;
    const int aggN = N * HD;

    // carve workspace
    char* p = (char*)d_ws;
    auto carve = [&](size_t bytes) -> void* {
        void* r = (void*)p;
        p += (bytes + 255) & ~(size_t)255;
        return r;
    };
    unsigned short* hb    = (unsigned short*)carve((size_t)ND * 2);
    float* h              = (float*)carve((size_t)ND * 4);
    float* qB             = (float*)carve((size_t)N * HD * 4);
    float* kB             = (float*)carve((size_t)N * HD * 4);
    float* vB             = (float*)carve((size_t)N * HD * 4);
    float* skipB          = (float*)carve((size_t)ND * 4);
    float* agg            = (float*)carve((size_t)N * HD * 4);
    float* scores         = (float*)carve((size_t)EH * 4);
    unsigned* m_enc       = (unsigned*)carve((size_t)NH * 4);
    float* denom          = (float*)carve((size_t)NH * 4);
    float* pooled         = (float*)carve((size_t)GG * DD * 4);
    float* cnt            = (float*)carve((size_t)GG * 4);
    unsigned short* Wqt   = (unsigned short*)carve((size_t)HD * DD * 2);
    unsigned short* Wkt   = (unsigned short*)carve((size_t)HD * DD * 2);
    unsigned short* Wvt   = (unsigned short*)carve((size_t)HD * DD * 2);
    unsigned short* Wst   = (unsigned short*)carve((size_t)DD * DD * 2);

    const int B = 256;

    // initial activations -> bf16
    cvt_f32_bf16_kernel<<<cdiv(ND, B), B, 0, stream>>>(x, hb, ND);

    // one wave per 16x64 strip, 8 waves per block
    auto gemm_grid = [](int M, int Nout) { return ((M / 16) * (Nout / 64) + 7) / 8; };

    for (int l = 0; l < LL; ++l) {
        const float* Wq_l = Wq + (size_t)l * DD * HD;
        const float* Wk_l = Wk + (size_t)l * DD * HD;
        const float* Wv_l = Wv + (size_t)l * DD * HD;
        const float* Ws_l = Wskip + (size_t)l * DD * DD;
        const float* bq_l = bq + (size_t)l * HD;
        const float* bk_l = bk + (size_t)l * HD;
        const float* bv_l = bv + (size_t)l * HD;
        const float* bs_l = bskip + (size_t)l * DD;

        // weights -> bf16 transposed [Nout, 128]
        cvt_transpose_w_kernel<<<cdiv(DD * HD, B), B, 0, stream>>>(Wq_l, Wqt, DD, HD);
        cvt_transpose_w_kernel<<<cdiv(DD * HD, B), B, 0, stream>>>(Wk_l, Wkt, DD, HD);
        cvt_transpose_w_kernel<<<cdiv(DD * HD, B), B, 0, stream>>>(Wv_l, Wvt, DD, HD);
        cvt_transpose_w_kernel<<<cdiv(DD * DD, B), B, 0, stream>>>(Ws_l, Wst, DD, DD);

        // zero/seed per-layer accumulators
        init_layer_kernel<<<cdiv(aggN, B), B, 0, stream>>>(m_enc, denom, agg, NH, aggN);

        // projections (WMMA bf16, fp32 accumulate)
        gemm_bf16_wmma_kernel<<<gemm_grid(N, HD), B, 0, stream>>>(hb, Wqt, bq_l, qB, N, HD);
        gemm_bf16_wmma_kernel<<<gemm_grid(N, HD), B, 0, stream>>>(hb, Wkt, bk_l, kB, N, HD);
        gemm_bf16_wmma_kernel<<<gemm_grid(N, HD), B, 0, stream>>>(hb, Wvt, bv_l, vB, N, HD);
        gemm_bf16_wmma_kernel<<<gemm_grid(N, DD), B, 0, stream>>>(hb, Wst, bs_l, skipB, N, DD);

        // edge attention
        edge_scores_kernel<<<cdiv((long long)E * 32, B), B, 0, stream>>>(qB, kB, src, dst, scores, E);
        edge_segmax_kernel<<<cdiv(EH, B), B, 0, stream>>>(scores, dst, m_enc, EH);
        edge_exp_kernel<<<cdiv(EH, B), B, 0, stream>>>(scores, dst, m_enc, denom, EH);
        edge_aggregate_kernel<<<cdiv((long long)EH * 32, B), B, 0, stream>>>(scores, denom, vB, src, dst, agg, E);

        // mean-heads + skip + relu -> next layer input
        finalize_layer_kernel<<<cdiv(ND, B), B, 0, stream>>>(agg, skipB, h, hb, ND);
    }

    // global mean pool + classifier
    pool_init_kernel<<<cdiv(GG * DD, B), B, 0, stream>>>(pooled, cnt);
    pool_kernel<<<cdiv(ND, B), B, 0, stream>>>(h, batch, pooled, cnt, ND);
    classifier_kernel<<<cdiv(GG * OUTD, B), B, 0, stream>>>(pooled, cnt, Wc, bc, out);
}
